// LovaszSoftmaxLoss_46076409151793
// MI455X (gfx1250) — matrix operand order repaired and checked by reading, not compiled, not run on hardware
//
#include <hip/hip_runtime.h>
#include <stdint.h>

#define NPTS  4000000
#define NCLS  13
#define SORT_ITEMS 4096
#define SORT_BLOCKS 977                 // 977*4096 = 4,001,792
#define NPAD  (SORT_BLOCKS * SORT_ITEMS)
#define SCAN_ITEMS 2048
#define SCAN_BLOCKS 1954                // 1954*2048 = 4,001,792

typedef __attribute__((ext_vector_type(16))) _Float16 v16h;
typedef __attribute__((ext_vector_type(8)))  _Float16 v8h;
typedef __attribute__((ext_vector_type(8)))  float    v8f;
typedef unsigned int u32;
typedef unsigned long long u64;

// ---------------------------------------------------------------------------
// Build sortable items for one class: key = ~bits(|fg - softmax_c|) in high 32
// (ascending u32 sort == descending error, stable), fg in bit 0.
// ---------------------------------------------------------------------------
__global__ void __launch_bounds__(256)
build_items(const float* __restrict__ logits,
            const int* __restrict__ targets,
            u64* __restrict__ out, int cls) {
  long i = (long)blockIdx.x * 256 + threadIdx.x;
  if (i >= NPAD) return;
  u64 item;
  if (i < NPTS) {
    const float* row = logits + i * NCLS;
    float m = row[0];
    #pragma unroll
    for (int c = 1; c < NCLS; ++c) m = fmaxf(m, row[c]);
    float s = 0.f, pc = 0.f;
    #pragma unroll
    for (int c = 0; c < NCLS; ++c) {
      float e = __expf(row[c] - m);
      s += e;
      if (c == cls) pc = e;
    }
    pc /= s;
    u32 fg = (targets[i] == cls) ? 1u : 0u;
    float err = fabsf((float)fg - pc);
    u32 key = ~__float_as_uint(err);
    item = ((u64)key << 32) | (u64)fg;
  } else {
    item = 0xFFFFFFFF00000000ull;       // pad: err = +0.0, fg = 0, sorts last
  }
  out[i] = item;
}

// ---------------------------------------------------------------------------
// Radix pass: per-block 16-bucket histogram
// ---------------------------------------------------------------------------
__global__ void __launch_bounds__(256)
radix_hist(const u64* __restrict__ keys, u32* __restrict__ hist, int shift) {
  __shared__ u32 cnt[16];
  int t = threadIdx.x;
  if (t < 16) cnt[t] = 0;
  __syncthreads();
  long base = (long)blockIdx.x * SORT_ITEMS + t * 16;
  #pragma unroll
  for (int q = 0; q < 16; ++q) {
    u32 d = (u32)(keys[base + q] >> shift) & 15u;
    atomicAdd(&cnt[d], 1u);
  }
  __syncthreads();
  if (t < 16) hist[blockIdx.x * 16 + t] = cnt[t];
}

// ---------------------------------------------------------------------------
// Radix pass: scan histograms.  hist[b][d] -> exclusive-over-blocks count for
// digit d; digitBase[d] -> exclusive-over-digits total.
// ---------------------------------------------------------------------------
__global__ void __launch_bounds__(256)
radix_scan_hist(u32* __restrict__ hist, u32* __restrict__ digitBase) {
  __shared__ u32 segSum[256];   // [seg*16 + d]
  __shared__ u32 segOff[256];
  __shared__ u32 digTot[16], digBase[16];
  const int SEG = (SORT_BLOCKS + 15) / 16;   // 62
  int t = threadIdx.x;
  int d = t & 15, seg = t >> 4;
  int b0 = seg * SEG;
  int b1 = (b0 + SEG < SORT_BLOCKS) ? (b0 + SEG) : SORT_BLOCKS;
  u32 s = 0;
  for (int b = b0; b < b1; ++b) s += hist[b * 16 + d];
  segSum[seg * 16 + d] = s;
  __syncthreads();
  if (t < 16) {
    u32 run = 0;
    for (int g = 0; g < 16; ++g) { u32 v = segSum[g * 16 + t]; segOff[g * 16 + t] = run; run += v; }
    digTot[t] = run;
  }
  __syncthreads();
  if (t == 0) {
    u32 run = 0;
    for (int dd = 0; dd < 16; ++dd) { digBase[dd] = run; run += digTot[dd]; }
  }
  __syncthreads();
  u32 run = segOff[seg * 16 + d];
  for (int b = b0; b < b1; ++b) { u32 v = hist[b * 16 + d]; hist[b * 16 + d] = run; run += v; }
  if (t < 16) digitBase[t] = digBase[t];
}

// ---------------------------------------------------------------------------
// Radix pass: stable scatter.  Block-local stable ranks via a digit-major
// flattened scan of per-(thread,digit) counts.
// ---------------------------------------------------------------------------
__global__ void __launch_bounds__(256)
radix_scatter(const u64* __restrict__ in, u64* __restrict__ out,
              const u32* __restrict__ hist,
              const u32* __restrict__ digitBase, int shift) {
  __shared__ u32 cnt[16 * 256];   // [d*256 + t]
  __shared__ u32 aux[256];
  __shared__ u32 below[16];
  __shared__ u32 dbase[16], bexc[16];
  int t = threadIdx.x;
  #pragma unroll
  for (int k = 0; k < 16; ++k) cnt[k * 256 + t] = 0;
  if (t < 16) { dbase[t] = digitBase[t]; bexc[t] = hist[blockIdx.x * 16 + t]; }
  __syncthreads();
  long base = (long)blockIdx.x * SORT_ITEMS + t * 16;   // blocked => stable order
  u64 it[16];
  #pragma unroll
  for (int q = 0; q < 16; ++q) {
    it[q] = in[base + q];
    u32 d = (u32)(it[q] >> shift) & 15u;
    cnt[d * 256 + t] += 1;                 // slot owned by this thread
  }
  __syncthreads();
  // exclusive scan of the 4096 flattened counters; thread t scans chunk [16t,16t+16)
  u32 run = 0;
  #pragma unroll
  for (int q = 0; q < 16; ++q) { u32 v = cnt[t * 16 + q]; cnt[t * 16 + q] = run; run += v; }
  aux[t] = run;
  __syncthreads();
  for (int ofs = 1; ofs < 256; ofs <<= 1) {      // Hillis-Steele inclusive
    u32 v = (t >= ofs) ? aux[t - ofs] : 0u;
    __syncthreads();
    aux[t] += v;
    __syncthreads();
  }
  u32 chunkBase = aux[t] - run;
  #pragma unroll
  for (int q = 0; q < 16; ++q) cnt[t * 16 + q] += chunkBase;
  __syncthreads();
  if (t < 16) below[t] = cnt[t * 256];    // sum of lower-digit counts in block
  __syncthreads();
  #pragma unroll
  for (int q = 0; q < 16; ++q) {
    u32 d = (u32)(it[q] >> shift) & 15u;
    u32 r = cnt[d * 256 + t]++;           // block-local flat rank
    u32 dest = dbase[d] + bexc[d] + (r - below[d]);
    out[dest] = it[q];
  }
}

// ---------------------------------------------------------------------------
// Per-block fg sums over the sorted array
// ---------------------------------------------------------------------------
__global__ void __launch_bounds__(256)
fg_block_sum(const u64* __restrict__ keys, float* __restrict__ blockSums) {
  __shared__ u32 red[256];
  int t = threadIdx.x;
  long base = (long)blockIdx.x * SCAN_ITEMS + t * 8;
  u32 s = 0;
  #pragma unroll
  for (int q = 0; q < 8; ++q) s += (u32)(keys[base + q] & 1ull);
  red[t] = s;
  __syncthreads();
  for (int ofs = 128; ofs > 0; ofs >>= 1) {
    if (t < ofs) red[t] += red[t + ofs];
    __syncthreads();
  }
  if (t == 0) blockSums[blockIdx.x] = (float)red[0];
}

// Exclusive scan of block sums (in place) + total G.  Counts <= 4M: exact in f32.
__global__ void __launch_bounds__(256)
scan_block_sums(float* __restrict__ blockSums, float* __restrict__ gOut) {
  __shared__ float aux[256];
  const int CH = (SCAN_BLOCKS + 255) / 256;   // 8
  int t = threadIdx.x;
  int b0 = t * CH;
  float loc[CH];
  float s = 0.f;
  #pragma unroll
  for (int q = 0; q < CH; ++q) {
    int b = b0 + q;
    float v = (b < SCAN_BLOCKS) ? blockSums[b] : 0.f;
    loc[q] = s; s += v;
  }
  aux[t] = s;
  __syncthreads();
  for (int ofs = 1; ofs < 256; ofs <<= 1) {
    float v = (t >= ofs) ? aux[t - ofs] : 0.f;
    __syncthreads();
    aux[t] += v;
    __syncthreads();
  }
  float chunkBase = aux[t] - s;
  #pragma unroll
  for (int q = 0; q < CH; ++q) {
    int b = b0 + q;
    if (b < SCAN_BLOCKS) blockSums[b] = chunkBase + loc[q];
  }
  if (t == 255) gOut[0] = aux[255];
}

// ---------------------------------------------------------------------------
// Fused scan + Lovasz-grad dot product.  Per-wave 256-element inclusive scan
// on the matrix core:  M1 = X*U (row prefixes),  D = S*RowTot + M1.
// 2x v_wmma_f32_16x16x32_f16 per 256-element chunk; counts exact in f16/f32.
// All LDS traffic is unconditional + vectorized (b128) to avoid predicated
// scalar ds loads between the WMMAs.
// ---------------------------------------------------------------------------
__global__ void __launch_bounds__(256)
scan_loss(const u64* __restrict__ keys,
          const float* __restrict__ blockOffs,
          const float* __restrict__ gTot,
          float* __restrict__ partials) {
  __shared__ __align__(16) _Float16 xlds[8 * 256];
  __shared__ __align__(16) _Float16 rowTotH[8 * 16];
  __shared__ float scanv[8 * 256];
  __shared__ float waveOff[8];
  __shared__ float red[256];
  int t = threadIdx.x;
  int w = t >> 5, lane = t & 31;
  long base = (long)blockIdx.x * SCAN_ITEMS;
  int cbase = w * 256 + lane * 8;

  u64 it[8];
  float ev[8]; u32 fg[8];
  v8h xstore;
  #pragma unroll
  for (int q = 0; q < 8; ++q) {
    it[q] = keys[base + cbase + q];
    fg[q] = (u32)(it[q] & 1ull);
    ev[q] = __uint_as_float(~(u32)(it[q] >> 32));
    xstore[q] = (_Float16)(float)fg[q];
  }
  *(v8h*)&xlds[cbase] = xstore;          // X row-major: idx = 16*row+col (one b128)
  __syncthreads();

  int col = lane & 15, hi = lane >> 4;
  // A fragment (16x32, K>=16 zero): lanes 0-15 carry K=0..7, lanes 16-31 K=8..15.
  // The 8 live halves are contiguous in LDS -> single ds_load_b128.
  v8h xv = *(const v8h*)&xlds[w * 256 + col * 16 + hi * 8];
  v16h a, b;
  #pragma unroll
  for (int e = 0; e < 16; ++e) a[e] = (e < 8) ? xv[e] : (_Float16)0.f;
  // B = U: upper-triangular (incl diag) ones; element e<8 of lane -> row k=e+8*hi (K>=16 zero)
  #pragma unroll
  for (int e = 0; e < 16; ++e)
    b[e] = (_Float16)((e < 8 && e + 8 * hi <= col) ? 1.f : 0.f);

  v8f m1 = {};
  m1 = __builtin_amdgcn_wmma_f32_16x16x32_f16(false, a, false, b, (short)0, m1,
                                              false, false);
  // Row totals live in column 15 (lanes 15 and 31); store as f16 (exact, <=16).
  if (col == 15) {
    v8h rt;
    #pragma unroll
    for (int r = 0; r < 8; ++r) rt[r] = (_Float16)m1[r];
    *(v8h*)&rowTotH[w * 16 + 8 * hi] = rt;
  }
  __syncthreads();

  // Unconditional vector loads of row totals; lane-select afterwards (cndmask).
  v8h rlo = *(const v8h*)&rowTotH[w * 16];
  v8h rhi = *(const v8h*)&rowTotH[w * 16 + 8];
  v16h a2, b2;
  #pragma unroll
  for (int e = 0; e < 16; ++e) {
    int K = e + ((e >= 8) ? 8 : 0) + hi * 8;
    a2[e] = (_Float16)((e < 8 && K < col) ? 1.f : 0.f);   // strict lower triangular
    _Float16 rv = (hi == 0) ? rlo[e & 7] : rhi[e & 7];
    b2[e] = (e < 8) ? rv : (_Float16)0.f;
  }
  v8f d2 = __builtin_amdgcn_wmma_f32_16x16x32_f16(false, a2, false, b2, (short)0,
                                                  m1, false, false);
  #pragma unroll
  for (int r = 0; r < 8; ++r)
    scanv[w * 256 + (r + 8 * hi) * 16 + col] = d2[r];  // inclusive scan, idx=16m+n
  __syncthreads();

  if (t == 0) {
    float run = 0.f;
    for (int ww = 0; ww < 8; ++ww) { waveOff[ww] = run; run += scanv[ww * 256 + 255]; }
  }
  __syncthreads();

  float G = gTot[0];
  float bOff = blockOffs[blockIdx.x];
  float acc = 0.f;
  #pragma unroll
  for (int q = 0; q < 8; ++q) {
    long gi = base + cbase + q;
    float F = bOff + waveOff[w] + scanv[cbase + q];     // inclusive fg count
    float fi = (float)gi;
    float jac1 = 1.f - (G - F) / (G + (fi + 1.f) - F);
    float Fp = F - (float)fg[q];
    float jac0 = (gi > 0) ? (1.f - (G - Fp) / (G + fi - Fp)) : 0.f;
    acc += ev[q] * (jac1 - jac0);
  }
  red[t] = acc;
  __syncthreads();
  for (int ofs = 128; ofs > 0; ofs >>= 1) {
    if (t < ofs) red[t] += red[t + ofs];
    __syncthreads();
  }
  if (t == 0) partials[blockIdx.x] = red[0];
}

// Deterministic fixed-order reduction of block partials -> per-class loss.
__global__ void __launch_bounds__(256)
reduce_partials(const float* __restrict__ partials,
                float* __restrict__ lossAccum, int cls) {
  __shared__ float red[256];
  int t = threadIdx.x;
  float s = 0.f;
  for (int i = t; i < SCAN_BLOCKS; i += 256) s += partials[i];
  red[t] = s;
  __syncthreads();
  for (int ofs = 128; ofs > 0; ofs >>= 1) {
    if (t < ofs) red[t] += red[t + ofs];
    __syncthreads();
  }
  if (t == 0) lossAccum[cls] = red[0];
}

__global__ void finalize_loss(const float* __restrict__ lossAccum, float* __restrict__ out) {
  if (threadIdx.x == 0 && blockIdx.x == 0) {
    float s = 0.f;
    for (int c = 0; c < NCLS; ++c) s += lossAccum[c];
    out[0] = s / (float)NCLS;
  }
}

// ---------------------------------------------------------------------------
extern "C" void kernel_launch(void* const* d_in, const int* in_sizes, int n_in,
                              void* d_out, int out_size, void* d_ws, size_t ws_size,
                              hipStream_t stream) {
  const float* logits = (const float*)d_in[0];
  const int* targets  = (const int*)d_in[1];
  float* out = (float*)d_out;

  char* ws = (char*)d_ws;
  size_t off = 0;
  auto alloc = [&](size_t bytes) {
    off = (off + 255) & ~(size_t)255;
    void* p = ws + off;
    off += bytes;
    return p;
  };
  u64* buf0        = (u64*)alloc((size_t)NPAD * 8);
  u64* buf1        = (u64*)alloc((size_t)NPAD * 8);
  u32* hist        = (u32*)alloc((size_t)SORT_BLOCKS * 16 * 4);
  u32* digitBase   = (u32*)alloc(64);
  float* blockSums = (float*)alloc((size_t)SCAN_BLOCKS * 4);
  float* gTot      = (float*)alloc(4);
  float* partials  = (float*)alloc((size_t)SCAN_BLOCKS * 4);
  float* lossAccum = (float*)alloc((size_t)NCLS * 4);

  for (int cls = 0; cls < NCLS; ++cls) {
    build_items<<<(NPAD + 255) / 256, 256, 0, stream>>>(logits, targets, buf0, cls);
    u64* src = buf0;
    u64* dst = buf1;
    for (int pass = 0; pass < 8; ++pass) {
      int shift = 32 + 4 * pass;
      radix_hist<<<SORT_BLOCKS, 256, 0, stream>>>(src, hist, shift);
      radix_scan_hist<<<1, 256, 0, stream>>>(hist, digitBase);
      radix_scatter<<<SORT_BLOCKS, 256, 0, stream>>>(src, dst, hist, digitBase, shift);
      u64* tmp = src; src = dst; dst = tmp;
    }
    // after an even number of passes the sorted items are back in buf0 (== src)
    fg_block_sum<<<SCAN_BLOCKS, 256, 0, stream>>>(src, blockSums);
    scan_block_sums<<<1, 256, 0, stream>>>(blockSums, gTot);
    scan_loss<<<SCAN_BLOCKS, 256, 0, stream>>>(src, blockSums, gTot, partials);
    reduce_partials<<<1, 256, 0, stream>>>(partials, lossAccum, cls);
  }
  finalize_loss<<<1, 1, 0, stream>>>(lossAccum, out);
}
